// StackedDRNN_47399259079390
// MI455X (gfx1250) — matrix operand order repaired and checked by reading, not compiled, not run on hardware
//
#include <hip/hip_runtime.h>
#include <hip/hip_bf16.h>
#include <math.h>

// ---------------------------------------------------------------------------
// Stacked dilated LSTM (T=1024, B=16, D=H=1024, rates 1/2/4) for gfx1250.
//  - bf16 WMMA (v_wmma_f32_16x16x32_bf16) for xw = x@W and recurrent h@U
//  - shared operands staged via global_load_async_to_lds_* (ASYNCcnt) with
//    double buffering; per-wave streaming operands software-pipelined in regs
//  - fp32 cell state; per-timestep launches give sequential ordering
// ---------------------------------------------------------------------------

typedef __attribute__((ext_vector_type(16))) __bf16 bf16x16;
typedef __attribute__((ext_vector_type(8)))  __bf16 bf16x8;
typedef __attribute__((ext_vector_type(8)))  float  f32x8;

#define SEQ_LEN 1024
#define BATCH   16
#define DIM     1024
#define FOURH   4096
#define MROWS   (SEQ_LEN * BATCH)   // 16384
#define BPAD    80                  // padded LDS stride per 32-K chunk row

// ---- CDNA5 async copy helpers (ASYNCcnt-tracked global->LDS DMA) ----------
__device__ inline uint32_t lds_off32(const void* p) {
    // flat LDS address: addr[31:0] == LDS byte offset (aperture in high bits)
    return (uint32_t)(uintptr_t)p;
}
__device__ inline void async_b128(uint32_t lds, const void* g) {
    asm volatile("global_load_async_to_lds_b128 %0, %1, off" :: "v"(lds), "v"(g) : "memory");
}
__device__ inline void async_b32(uint32_t lds, const void* g) {
    asm volatile("global_load_async_to_lds_b32 %0, %1, off" :: "v"(lds), "v"(g) : "memory");
}
__device__ inline void wait_async0() {
    asm volatile("s_wait_asynccnt 0" ::: "memory");
}

// ---- fragment loaders (CDNA5 16-bit WMMA VGPR layouts) --------------------
// A (16x32 MxK): lane m=lane%16; half kh: elems 0..7 = K[kh*8..], 8..15 = K[16+kh*8..]
__device__ inline bf16x16 frag_a_global(const __bf16* __restrict__ row, int k0, int kh) {
    union { bf16x16 v; bf16x8 h[2]; } u;
    u.h[0] = *(const bf16x8*)(row + k0 + kh * 8);
    u.h[1] = *(const bf16x8*)(row + k0 + 16 + kh * 8);
    return u.v;
}
// B (32x16 KxN): lane n=lane%16; half kh: elems 0..7 = K[kh*8..], 8..15 = K[16+kh*8..]
__device__ inline bf16x16 frag_b_global(const __bf16* __restrict__ col, int k0, int kh) {
    union { bf16x16 v; bf16x8 h[2]; } u;
    u.h[0] = *(const bf16x8*)(col + k0 + kh * 8);
    u.h[1] = *(const bf16x8*)(col + k0 + 16 + kh * 8);
    return u.v;
}
// generic LDS fragment: two 16B halves at byte offsets o0/o1
__device__ inline bf16x16 frag_lds(const char* p, int o0, int o1) {
    union { bf16x16 v; bf16x8 h[2]; } u;
    u.h[0] = *(const bf16x8*)(p + o0);
    u.h[1] = *(const bf16x8*)(p + o1);
    return u.v;
}

__device__ inline float sigmoid_f(float x) { return 1.0f / (1.0f + __expf(-x)); }

// ---- fp32 -> bf16 conversion of input sequence ----------------------------
__global__ __launch_bounds__(256) void convert_x_kernel(
    const float* __restrict__ src, __bf16* __restrict__ dst) {
    int idx = blockIdx.x * 256 + threadIdx.x;
    dst[idx] = (__bf16)src[idx];
}

// ---- fuse 4 gate weights into transposed bf16 [4H][D] (K contiguous) ------
__global__ __launch_bounds__(256) void convert_w_kernel(
    const float* __restrict__ wi, const float* __restrict__ wf,
    const float* __restrict__ wo, const float* __restrict__ wg,
    __bf16* __restrict__ dstT) {
    int idx = blockIdx.x * 256 + threadIdx.x;
    int n = idx >> 10;
    int k = idx & 1023;
    int gate = n >> 10;
    int hcol = n & 1023;
    const float* w = (gate == 0) ? wi : (gate == 1) ? wf : (gate == 2) ? wo : wg;
    dstT[idx] = (__bf16)w[k * DIM + hcol];
}

// ---- input projection: xw = act @ W + b ; M=16384 N=4096 K=1024 -----------
// Block (8 waves) computes 128 rows x 64 cols. B chunk (64 cols x 32 K, 4KB)
// is shared by all waves: async-staged to LDS, double buffered. A fragments
// are per-wave, register double-buffered.
__global__ __launch_bounds__(256) void gemm_xw_kernel(
    const __bf16* __restrict__ A,    // [MROWS][DIM]
    const __bf16* __restrict__ Bt,   // [FOURH][DIM]
    const float* __restrict__ bi, const float* __restrict__ bf_,
    const float* __restrict__ bo, const float* __restrict__ bg,
    __bf16* __restrict__ Out) {      // [MROWS][FOURH]
    __shared__ char Bsh[2][64 * BPAD];          // 2 x 5KB
    const int tid  = threadIdx.x;
    const int wv   = tid >> 5;
    const int lane = tid & 31;
    const int lm   = lane & 15;
    const int kh   = lane >> 4;
    const int mblk = blockIdx.x >> 6;           // 0..127
    const int n0   = (blockIdx.x & 63) * 64;
    const int mtile = mblk * 8 + wv;

    const __bf16* arow = A + (size_t)(mtile * 16 + lm) * DIM;

    // staging plan: thread copies 16B of the 64x32K chunk
    const int scol = tid >> 2, spart = tid & 3;
    const __bf16* bsrc = Bt + (size_t)(n0 + scol) * DIM + spart * 8;
    const uint32_t ldst0 = lds_off32(&Bsh[0][scol * BPAD + spart * 16]);
    const uint32_t ldst1 = lds_off32(&Bsh[1][scol * BPAD + spart * 16]);

    async_b128(ldst0, bsrc);                    // k0 = 0 chunk

    f32x8 acc0 = {}, acc1 = {}, acc2 = {}, acc3 = {};
    bf16x16 a = frag_a_global(arow, 0, kh);
    for (int k0 = 0; k0 < DIM; k0 += 32) {
        const int cur = (k0 >> 5) & 1;
        wait_async0();
        __syncthreads();                        // chunk[cur] visible to all waves
        if (k0 + 32 < DIM) {                    // overlap next DMA with compute
            async_b128(cur ? ldst0 : ldst1, bsrc + k0 + 32);
            __builtin_prefetch(arow + k0 + 128, 0, 1);
        }
        bf16x16 an = a;
        if (k0 + 32 < DIM) an = frag_a_global(arow, k0 + 32, kh);
        const char* bb = &Bsh[cur][0];
        bf16x16 b0 = frag_lds(bb, (0 * 16 + lm) * BPAD + kh * 16, (0 * 16 + lm) * BPAD + 32 + kh * 16);
        bf16x16 b1 = frag_lds(bb, (1 * 16 + lm) * BPAD + kh * 16, (1 * 16 + lm) * BPAD + 32 + kh * 16);
        bf16x16 b2 = frag_lds(bb, (2 * 16 + lm) * BPAD + kh * 16, (2 * 16 + lm) * BPAD + 32 + kh * 16);
        bf16x16 b3 = frag_lds(bb, (3 * 16 + lm) * BPAD + kh * 16, (3 * 16 + lm) * BPAD + 32 + kh * 16);
        acc0 = __builtin_amdgcn_wmma_f32_16x16x32_bf16(false, a, false, b0, (short)0, acc0, false, false);
        acc1 = __builtin_amdgcn_wmma_f32_16x16x32_bf16(false, a, false, b1, (short)0, acc1, false, false);
        acc2 = __builtin_amdgcn_wmma_f32_16x16x32_bf16(false, a, false, b2, (short)0, acc2, false, false);
        acc3 = __builtin_amdgcn_wmma_f32_16x16x32_bf16(false, a, false, b3, (short)0, acc3, false, false);
        a = an;
    }

    const int gate = n0 >> 10;                  // 64-col slab is gate-uniform
    const float* bp = (gate == 0) ? bi : (gate == 1) ? bf_ : (gate == 2) ? bo : bg;
    f32x8 accs[4] = {acc0, acc1, acc2, acc3};
#pragma unroll
    for (int c = 0; c < 4; ++c) {
        int n = n0 + c * 16 + lm;
        float bv = bp[n & 1023];
#pragma unroll
        for (int r = 0; r < 8; ++r) {
            int m = mtile * 16 + r + 8 * kh;    // C/D layout: M = r + 8*kh
            Out[(size_t)m * FOURH + n] = (__bf16)(accs[c][r] + bv);
        }
    }
}

// ---- one dilated-LSTM timestep: gates = xw_t + h_prev @ U ; cell ; h ------
// All 8 waves of a block share the same 16 A-rows (h_prev) -> async-stage A
// to LDS, double buffered. Per-wave U columns are register double-buffered.
__global__ __launch_bounds__(256) void lstm_step_kernel(
    const __bf16* __restrict__ XW,   // [MROWS][FOURH]
    const __bf16* __restrict__ Ut,   // [FOURH][DIM]
    __bf16* __restrict__ Act,        // [MROWS][DIM]
    float* __restrict__ Cst,         // [64][DIM]
    float* __restrict__ OutF,        // fp32 h output (last layer) or null
    int s, int rate) {
    __shared__ char Ash[2][16 * BPAD];          // 2 x 1.25KB
    const int tid   = threadIdx.x;
    const int wv    = tid >> 5;
    const int lane  = tid & 31;
    const int gwave = blockIdx.x * 8 + wv;      // 0..rate*64-1
    const int M     = rate * 16;
    const int mtile = gwave >> 6;               // block-uniform
    const int htile = gwave & 63;
    const int lm    = lane & 15;
    const int kh    = lane >> 4;
    const int rowbase = s * M;

    f32x8 acc[4];
#pragma unroll
    for (int g = 0; g < 4; ++g) {
        int n = g * 1024 + htile * 16 + lm;
#pragma unroll
        for (int r = 0; r < 8; ++r) {
            int m = rowbase + mtile * 16 + r + 8 * kh;
            acc[g][r] = (float)XW[(size_t)m * FOURH + n];
        }
    }

    if (s > 0) {
        // staging plan: thread copies 4B of the 16x32K A chunk (1KB)
        const int srow = tid >> 4, spart = tid & 15;
        const __bf16* asrc = Act + (size_t)((s - 1) * M + mtile * 16 + srow) * DIM + spart * 2;
        const uint32_t ldst0 = lds_off32(&Ash[0][srow * BPAD + spart * 4]);
        const uint32_t ldst1 = lds_off32(&Ash[1][srow * BPAD + spart * 4]);
        async_b32(ldst0, asrc);

        const __bf16* b0 = Ut + (size_t)(0 * 1024 + htile * 16 + lm) * DIM;
        const __bf16* b1 = Ut + (size_t)(1 * 1024 + htile * 16 + lm) * DIM;
        const __bf16* b2 = Ut + (size_t)(2 * 1024 + htile * 16 + lm) * DIM;
        const __bf16* b3 = Ut + (size_t)(3 * 1024 + htile * 16 + lm) * DIM;
        bf16x16 f0 = frag_b_global(b0, 0, kh);
        bf16x16 f1 = frag_b_global(b1, 0, kh);
        bf16x16 f2 = frag_b_global(b2, 0, kh);
        bf16x16 f3 = frag_b_global(b3, 0, kh);

        for (int k0 = 0; k0 < DIM; k0 += 32) {
            const int cur = (k0 >> 5) & 1;
            wait_async0();
            __syncthreads();                    // A chunk[cur] visible
            if (k0 + 32 < DIM) async_b32(cur ? ldst0 : ldst1, asrc + k0 + 32);
            const char* ab = &Ash[cur][0];
            bf16x16 a = frag_lds(ab, lm * BPAD + kh * 16, lm * BPAD + 32 + kh * 16);
            bf16x16 n0f = f0, n1f = f1, n2f = f2, n3f = f3;
            if (k0 + 32 < DIM) {                // pipeline next U fragments
                n0f = frag_b_global(b0, k0 + 32, kh);
                n1f = frag_b_global(b1, k0 + 32, kh);
                n2f = frag_b_global(b2, k0 + 32, kh);
                n3f = frag_b_global(b3, k0 + 32, kh);
            }
            acc[0] = __builtin_amdgcn_wmma_f32_16x16x32_bf16(false, a, false, f0, (short)0, acc[0], false, false);
            acc[1] = __builtin_amdgcn_wmma_f32_16x16x32_bf16(false, a, false, f1, (short)0, acc[1], false, false);
            acc[2] = __builtin_amdgcn_wmma_f32_16x16x32_bf16(false, a, false, f2, (short)0, acc[2], false, false);
            acc[3] = __builtin_amdgcn_wmma_f32_16x16x32_bf16(false, a, false, f3, (short)0, acc[3], false, false);
            f0 = n0f; f1 = n1f; f2 = n2f; f3 = n3f;
        }
    }

#pragma unroll
    for (int r = 0; r < 8; ++r) {
        float iv = sigmoid_f(acc[0][r]);
        float fv = sigmoid_f(acc[1][r]);
        float ov = sigmoid_f(acc[2][r]);
        float gv = tanhf(acc[3][r]);
        int lrow = mtile * 16 + r + 8 * kh;
        int col  = htile * 16 + lm;
        int cidx = lrow * DIM + col;
        float cold = (s == 0) ? 0.0f : Cst[cidx];
        float cn = fv * cold + iv * gv;
        float hv = ov * tanhf(cn);
        Cst[cidx] = cn;
        size_t grow = (size_t)(rowbase + lrow) * DIM + col;
        Act[grow] = (__bf16)hv;
        if (OutF) OutF[grow] = hv;
    }
}

// ---------------------------------------------------------------------------
extern "C" void kernel_launch(void* const* d_in, const int* in_sizes, int n_in,
                              void* d_out, int out_size, void* d_ws, size_t ws_size,
                              hipStream_t stream) {
    (void)in_sizes; (void)n_in; (void)out_size; (void)ws_size;
    const float* x = (const float*)d_in[0];
    const float* W4[4] = {(const float*)d_in[1], (const float*)d_in[2],
                          (const float*)d_in[3], (const float*)d_in[4]};
    const float* U4[4] = {(const float*)d_in[5], (const float*)d_in[6],
                          (const float*)d_in[7], (const float*)d_in[8]};
    const float* B4[4] = {(const float*)d_in[9], (const float*)d_in[10],
                          (const float*)d_in[11], (const float*)d_in[12]};
    float* out = (float*)d_out;

    char* ws = (char*)d_ws;
    const size_t XW_BYTES  = (size_t)MROWS * FOURH * sizeof(__bf16);  // 128 MB
    const size_t ACT_BYTES = (size_t)MROWS * DIM * sizeof(__bf16);    //  32 MB
    const size_t WT_BYTES  = (size_t)FOURH * DIM * sizeof(__bf16);    //   8 MB
    __bf16* xw  = (__bf16*)(ws);
    __bf16* act = (__bf16*)(ws + XW_BYTES);
    __bf16* Wt  = (__bf16*)(ws + XW_BYTES + ACT_BYTES);
    __bf16* Ut  = (__bf16*)(ws + XW_BYTES + ACT_BYTES + WT_BYTES);
    float*  Cst = (float*)(ws + XW_BYTES + ACT_BYTES + 2 * WT_BYTES); // 256 KB

    convert_x_kernel<<<MROWS * DIM / 256, 256, 0, stream>>>(x, act);

    const int rates[3] = {1, 2, 4};
    const size_t WOFF = (size_t)DIM * DIM;
    for (int l = 0; l < 3; ++l) {
        convert_w_kernel<<<FOURH * DIM / 256, 256, 0, stream>>>(
            W4[0] + l * WOFF, W4[1] + l * WOFF, W4[2] + l * WOFF, W4[3] + l * WOFF, Wt);
        convert_w_kernel<<<FOURH * DIM / 256, 256, 0, stream>>>(
            U4[0] + l * WOFF, U4[1] + l * WOFF, U4[2] + l * WOFF, U4[3] + l * WOFF, Ut);
        gemm_xw_kernel<<<8192, 256, 0, stream>>>(
            act, Wt, B4[0] + l * DIM, B4[1] + l * DIM, B4[2] + l * DIM, B4[3] + l * DIM, xw);

        const int rate = rates[l];
        const int steps = SEQ_LEN / rate;
        float* outF = (l == 2) ? out : nullptr;
        for (int s = 0; s < steps; ++s) {
            lstm_step_kernel<<<rate * 8, 256, 0, stream>>>(xw, Ut, act, Cst, outF, s, rate);
        }
    }
}
